// KLLogRegLossColBERT_74466142978308
// MI455X (gfx1250) — compile-verified, matrix operand order checked by reading, not run on hardware
//
#include <hip/hip_runtime.h>
#include <hip/hip_bf16.h>

typedef __attribute__((ext_vector_type(16))) _Float16 v16h;
typedef __attribute__((ext_vector_type(8)))  _Float16 v8h;
typedef __attribute__((ext_vector_type(4)))  _Float16 v4h;
typedef __attribute__((ext_vector_type(8)))  float    v8f;

#define CB_B    64
#define CB_LQ   32
#define CB_LD   256
#define CB_D    128
#define CB_NWAY 8
#define CB_NEG  (-9999.0f)
#define CB_BETA 0.1f

// ---------------------------------------------------------------------------
// Query pre-pass: row-wise L2 normalization fp32 -> f16 (one wave32 per row).
// Only 2048 rows (1 MB read / 0.5 MB write); result stays L2-resident and is
// reused by all NWAY=8 doc slates of each batch row.
// ---------------------------------------------------------------------------
__global__ __launch_bounds__(256) void cb_qnorm_f16(const float* __restrict__ in,
                                                    _Float16* __restrict__ out,
                                                    int nrows) {
  int row  = blockIdx.x * 8 + (threadIdx.x >> 5);
  if (row >= nrows) return;
  int lane = threadIdx.x & 31;

  float4 v = *((const float4*)(in + (size_t)row * CB_D) + lane);
  float ss = v.x * v.x + v.y * v.y + v.z * v.z + v.w * v.w;
  #pragma unroll
  for (int off = 1; off < 32; off <<= 1) ss += __shfl_xor(ss, off, 32);
  float inv = 1.0f / fmaxf(sqrtf(ss), 1e-12f);

  v4h o;
  o[0] = (_Float16)(v.x * inv);
  o[1] = (_Float16)(v.y * inv);
  o[2] = (_Float16)(v.z * inv);
  o[3] = (_Float16)(v.w * inv);
  *(v4h*)(out + (size_t)row * CB_D + lane * 4) = o;
}

// ---------------------------------------------------------------------------
// Fragment gather for the CDNA5 16-bit 16x32 A/B layout from a row-major
// [*,128] f16 matrix: lane (r,hi) holds K = {k0+hi*8 .. +7, k0+16+hi*8 .. +7}.
// Two 16-byte loads per lane (global_load_b128 or ds_load_b128).
// ---------------------------------------------------------------------------
__device__ __forceinline__ v16h cb_load_frag(const _Float16* tile, int lane, int k0) {
  int r = lane & 15, hi = lane >> 4;
  const _Float16* p = tile + r * CB_D + k0 + hi * 8;
  v8h lo  = *(const v8h*)(p);
  v8h hi8 = *(const v8h*)(p + 16);
  v16h out;
  #pragma unroll
  for (int i = 0; i < 8; ++i) { out[i] = lo[i]; out[i + 8] = hi8[i]; }
  return out;
}

// ---------------------------------------------------------------------------
// Fused kernel: one block per (b,n).
//   Stage 1: read 256x128 fp32 doc tile ONCE from HBM; per-row sumsq via
//            shfl reduce; store raw f16 tile to LDS + dinv[k] = 1/||d_k||.
//   Stage 2: S = Qn(32x128) x Draw^T(128x256) via WMMA f16 (A from L2-resident
//            qn, B from LDS), then scale by dinv[k], mask, max over k, sum
//            over q -> scores[bn].
// 8 waves; wave w owns N-tiles {2w,2w+1} x M-tiles {0,1}.
// ---------------------------------------------------------------------------
__global__ __launch_bounds__(256) void cb_scores_fused(const _Float16* __restrict__ qn,
                                                       const float* __restrict__ docs,
                                                       const int* __restrict__ dmask,
                                                       float* __restrict__ scores) {
  __shared__ _Float16 ldsTile[CB_LD * CB_D];   // 64 KB raw f16 doc tile
  __shared__ float    ldsDinv[CB_LD];          // 1 KB per-row inverse norms
  __shared__ float    smax[8][32];             // cross-wave q-max exchange

  int bn = blockIdx.x;                 // 0..511
  int b  = bn / CB_NWAY;
  const _Float16* Q  = qn + (size_t)b * CB_LQ * CB_D;
  const float*    Dg = docs + (size_t)bn * CB_LD * CB_D;
  const int*      mk = dmask + (size_t)bn * CB_LD;

  int tid  = threadIdx.x;
  int w    = tid >> 5;
  int lane = tid & 31;
  int r    = lane & 15;
  int hi   = lane >> 4;

  // ---- Stage 1: doc tile load + norm (wave w handles rows w*32..w*32+31) ---
  for (int it = 0; it < 32; ++it) {
    int row = w * 32 + it;
    const float* rp = Dg + (size_t)row * CB_D;
    if (it + 1 < 32) __builtin_prefetch(rp + CB_D, 0, 1);  // global_prefetch_b8
    float4 v = *((const float4*)rp + lane);
    float ss = v.x * v.x + v.y * v.y + v.z * v.z + v.w * v.w;
    #pragma unroll
    for (int off = 1; off < 32; off <<= 1) ss += __shfl_xor(ss, off, 32);

    v4h o;
    o[0] = (_Float16)v.x; o[1] = (_Float16)v.y;
    o[2] = (_Float16)v.z; o[3] = (_Float16)v.w;
    *(v4h*)(ldsTile + (size_t)row * CB_D + lane * 4) = o;
    if (lane == 0) ldsDinv[row] = 1.0f / fmaxf(sqrtf(ss), 1e-12f);
  }
  __syncthreads();

  // ---- Stage 2: WMMA GEMM + masked max-reduce ------------------------------
  // Hoist A fragments (normalized query) once: reused for both N-tiles.
  v16h aq[2][4];
  #pragma unroll
  for (int ki = 0; ki < 4; ++ki) {
    aq[0][ki] = cb_load_frag(Q,             lane, ki * 32);  // q rows 0..15
    aq[1][ki] = cb_load_frag(Q + 16 * CB_D, lane, ki * 32);  // q rows 16..31
  }

  float runmax[2][8];
  #pragma unroll
  for (int m = 0; m < 2; ++m)
    #pragma unroll
    for (int i = 0; i < 8; ++i) runmax[m][i] = CB_NEG;

  #pragma unroll
  for (int nt = 0; nt < 2; ++nt) {
    int ntile = w * 2 + nt;                       // 0..15
    const _Float16* Bt = ldsTile + (size_t)ntile * 16 * CB_D;
    int   kg    = ntile * 16 + r;                 // this lane's k column
    float dinvk = ldsDinv[kg];
    bool  valid = (mk[kg] != 0);

    v8f acc0 = {};
    v8f acc1 = {};
    #pragma unroll
    for (int ki = 0; ki < 4; ++ki) {
      v16h bf = cb_load_frag(Bt, lane, ki * 32);  // ds_load_b128 x2
      acc0 = __builtin_amdgcn_wmma_f32_16x16x32_f16(false, aq[0][ki], false, bf,
                                                    (short)0, acc0, false, false);
      acc1 = __builtin_amdgcn_wmma_f32_16x16x32_f16(false, aq[1][ki], false, bf,
                                                    (short)0, acc1, false, false);
    }

    #pragma unroll
    for (int m = 0; m < 2; ++m) {
      #pragma unroll
      for (int i = 0; i < 8; ++i) {
        float v = valid ? (m == 0 ? acc0[i] : acc1[i]) * dinvk : CB_NEG;
        // max over the 16 columns held by this half-wave (xor<16 stays in half)
        #pragma unroll
        for (int off = 1; off < 16; off <<= 1) v = fmaxf(v, __shfl_xor(v, off, 32));
        runmax[m][i] = fmaxf(runmax[m][i], v);
      }
    }
  }

  // lanes 0 and 16 publish per-wave partial maxima: q = m*16 + hi*8 + i
  if (r == 0) {
    #pragma unroll
    for (int m = 0; m < 2; ++m)
      #pragma unroll
      for (int i = 0; i < 8; ++i)
        smax[w][m * 16 + hi * 8 + i] = runmax[m][i];
  }
  __syncthreads();

  if (tid < 32) {
    float mx = -1e30f;
    #pragma unroll
    for (int ww = 0; ww < 8; ++ww) mx = fmaxf(mx, smax[ww][tid]);
    #pragma unroll
    for (int off = 1; off < 32; off <<= 1) mx += __shfl_xor(mx, off, 32);
    if (tid == 0) scores[bn] = mx;
  }
}

// ---------------------------------------------------------------------------
// Epilogue: log_softmax over NWAY, KL vs teacher + BETA * positive CE.
// ---------------------------------------------------------------------------
__global__ __launch_bounds__(64) void cb_final_loss(const float* __restrict__ scores,
                                                    const float* __restrict__ labels,
                                                    float* __restrict__ out) {
  int b = threadIdx.x;                 // 0..63
  const float* s   = scores + b * CB_NWAY;
  const float* lab = labels + b * (2 * CB_NWAY);

  float mx = -1e30f;
  #pragma unroll
  for (int n = 0; n < CB_NWAY; ++n) mx = fmaxf(mx, s[n]);
  float se = 0.0f;
  #pragma unroll
  for (int n = 0; n < CB_NWAY; ++n) se += __expf(s[n] - mx);
  float lse = mx + __logf(se);

  float loss_kl = 0.0f, loss_pos = 0.0f;
  #pragma unroll
  for (int n = 0; n < CB_NWAY; ++n) {
    float ls = s[n] - lse;
    float t  = lab[n];
    loss_kl  += __expf(t) * (t - ls);
    loss_pos += -ls * lab[CB_NWAY + n];
  }
  float total = loss_kl * (1.0f / (CB_B * CB_NWAY)) + CB_BETA * loss_pos;

  #pragma unroll
  for (int off = 1; off < 32; off <<= 1) total += __shfl_xor(total, off, 32);
  __shared__ float sh[2];
  if ((threadIdx.x & 31) == 0) sh[threadIdx.x >> 5] = total;
  __syncthreads();
  if (threadIdx.x == 0) out[0] = sh[0] + sh[1];
}

// ---------------------------------------------------------------------------
extern "C" void kernel_launch(void* const* d_in, const int* in_sizes, int n_in,
                              void* d_out, int out_size, void* d_ws, size_t ws_size,
                              hipStream_t stream) {
  const float* query = (const float*)d_in[0];   // (64,32,128)
  const float* docs  = (const float*)d_in[1];   // (64,8,256,128)
  const int*   dmask = (const int*)d_in[2];     // (64,8,256)
  const float* labs  = (const float*)d_in[3];   // (64,16)
  float* out = (float*)d_out;

  const int qrows = CB_B * CB_LQ;               // 2048

  _Float16* qn = (_Float16*)d_ws;                       // 512 KB
  float* scores = (float*)(qn + (size_t)qrows * CB_D);  // 512 floats

  cb_qnorm_f16<<<(qrows + 7) / 8, 256, 0, stream>>>(query, qn, qrows);
  cb_scores_fused<<<CB_B * CB_NWAY, 256, 0, stream>>>(qn, docs, dmask, scores);
  cb_final_loss<<<1, 64, 0, stream>>>(scores, labs, out);
}
